// MGCN_55671366091268
// MI455X (gfx1250) — matrix operand order repaired and checked by reading, not compiled, not run on hardware
//
#include <hip/hip_runtime.h>
#include <hip/hip_bf16.h>

typedef __attribute__((ext_vector_type(16))) _Float16 v16h;
typedef __attribute__((ext_vector_type(8)))  float    v8f;

#define SLOPE 0.2f

// ---------------------------------------------------------------------------
// float atomic max via signed/unsigned integer ordering trick
// ---------------------------------------------------------------------------
__device__ __forceinline__ void atomicMaxF(float* addr, float val) {
    if (val >= 0.0f) {
        atomicMax((int*)addr, __float_as_int(val));
    } else {
        atomicMin((unsigned int*)addr, (unsigned int)__float_as_int(val));
    }
}

// ---------------------------------------------------------------------------
// Pack a row-major f32 weight matrix W[64 x Ncols] into WMMA B-fragment
// layout, f16:  Bp[((nt*2 + kstep)*32 + lane)*16 + e]
//   = (f16) W[(kstep*32 + (lane>=16)*16 + e) * Ncols + nt*16 + (lane&15)]
// One thread per (nt, kstep, lane); each writes one 32-byte lane chunk.
// ---------------------------------------------------------------------------
__global__ void mgcn_pack_b(const float* __restrict__ W, _Float16* __restrict__ Bp,
                            int Ncols) {
    int i = blockIdx.x * blockDim.x + threadIdx.x;
    int total = (Ncols >> 4) * 2 * 32;
    if (i >= total) return;
    int lane  = i & 31;
    int kstep = (i >> 5) & 1;
    int nt    = i >> 6;
    int col   = (nt << 4) + (lane & 15);
    int kbase = kstep * 32 + ((lane >> 4) << 4);
    _Float16* out = Bp + (size_t)i * 16;
#pragma unroll
    for (int e = 0; e < 16; ++e)
        out[e] = (_Float16)W[(size_t)(kbase + e) * Ncols + col];
}

// ---------------------------------------------------------------------------
// WMMA GEMM:  C[M x Ncols] (+)= A[M x 64] * B[64 x Ncols]
// A: row-major f32 (converted to f16 in-register, float4 loads)
// B: pre-packed f16 fragments (two b128 loads per K-step)
// One wave per 16x16 tile; M assumed a multiple of 16 (N = 50000 = 3125*16),
// so the epilogue is branch-free.
// ---------------------------------------------------------------------------
__global__ void mgcn_gemm_wmma(const float* __restrict__ A,
                               const _Float16* __restrict__ Bp,
                               float* __restrict__ C,
                               int M, int Ncols, int accumulate) {
    int wave = (int)((blockIdx.x * blockDim.x + threadIdx.x) >> 5);
    int lane = (int)(threadIdx.x & 31);
    int ntiles = Ncols >> 4;
    int mt = wave / ntiles;
    int nt = wave - mt * ntiles;
    int m0 = mt << 4;
    int n0 = nt << 4;
    if (m0 >= M) return;                 // uniform per wave

    int halfSel = lane >> 4;             // 0 or 1
    int lm      = lane & 15;
    const float* arow = A + (size_t)(m0 + lm) * 64;
    float* ccol = C + (size_t)(m0 + halfSel * 8) * Ncols + n0 + lm;

    v8f acc = {};
    if (accumulate) {
#pragma unroll
        for (int v = 0; v < 8; ++v) acc[v] = ccol[(size_t)v * Ncols];
    }

#pragma unroll
    for (int kk = 0; kk < 2; ++kk) {
        int kb = kk * 32 + halfSel * 16;
        float4 a0 = *(const float4*)(arow + kb);
        float4 a1 = *(const float4*)(arow + kb + 4);
        float4 a2 = *(const float4*)(arow + kb + 8);
        float4 a3 = *(const float4*)(arow + kb + 12);
        v16h af;
        af[0]  = (_Float16)a0.x; af[1]  = (_Float16)a0.y;
        af[2]  = (_Float16)a0.z; af[3]  = (_Float16)a0.w;
        af[4]  = (_Float16)a1.x; af[5]  = (_Float16)a1.y;
        af[6]  = (_Float16)a1.z; af[7]  = (_Float16)a1.w;
        af[8]  = (_Float16)a2.x; af[9]  = (_Float16)a2.y;
        af[10] = (_Float16)a2.z; af[11] = (_Float16)a2.w;
        af[12] = (_Float16)a3.x; af[13] = (_Float16)a3.y;
        af[14] = (_Float16)a3.z; af[15] = (_Float16)a3.w;

        v16h bf = *(const v16h*)(Bp + ((size_t)(nt * 2 + kk) * 32 + lane) * 16);

        acc = __builtin_amdgcn_wmma_f32_16x16x32_f16(
            /*neg_a=*/false, af, /*neg_b=*/false, bf,
            /*c_mod=*/(short)0, acc, /*reuse_a=*/false, /*reuse_b=*/false);
    }

#pragma unroll
    for (int v = 0; v < 8; ++v) ccol[(size_t)v * Ncols] = acc[v];
}

// ---------------------------------------------------------------------------
// el[n,h] = sum_d h[n,h,d]*al[h,d] ; er likewise. One thread per (n, head).
// ---------------------------------------------------------------------------
__global__ void mgcn_el_er(const float* __restrict__ h,
                           const float* __restrict__ al,
                           const float* __restrict__ ar,
                           float* __restrict__ el, float* __restrict__ er, int N) {
    int i = blockIdx.x * blockDim.x + threadIdx.x;
    if (i >= N * 2) return;
    int n = i >> 1, hd = i & 1;
    const float* hp  = h  + (size_t)n * 128 + hd * 64;
    const float* alp = al + hd * 64;
    const float* arp = ar + hd * 64;
    float sl = 0.f, sr = 0.f;
#pragma unroll 8
    for (int d = 0; d < 64; ++d) { float v = hp[d]; sl += v * alp[d]; sr += v * arp[d]; }
    el[i] = sl; er[i] = sr;
}

// ---------------------------------------------------------------------------
// init agg = 0 (N*128), m = -inf (N*2), denom = 0 (N*2)
// ---------------------------------------------------------------------------
__global__ void mgcn_init(float* __restrict__ agg, float* __restrict__ m,
                          float* __restrict__ denom, int N) {
    int i = blockIdx.x * blockDim.x + threadIdx.x;
    int tot = N * 128;
    if (i < tot) agg[i] = 0.f;
    if (i < N * 2) { m[i] = -INFINITY; denom[i] = 0.f; }
}

// ---------------------------------------------------------------------------
// pass 1: e = leaky_relu(el[src] + er[dst]) ; segment max into m[dst]
// ---------------------------------------------------------------------------
__global__ void mgcn_edge_max(const float* __restrict__ el, const float* __restrict__ er,
                              const int* __restrict__ src, const int* __restrict__ dst,
                              float* __restrict__ ebuf, float* __restrict__ m, int E) {
    int i = blockIdx.x * blockDim.x + threadIdx.x;
    if (i >= E * 2) return;
    int e = i >> 1, hd = i & 1;
    int s = src[e], d = dst[e];
    float v = el[s * 2 + hd] + er[d * 2 + hd];
    v = (v > 0.f) ? v : SLOPE * v;
    ebuf[i] = v;
    atomicMaxF(&m[d * 2 + hd], v);
}

// ---------------------------------------------------------------------------
// pass 2: ex = exp(e - m[dst]) in place ; segment sum into denom[dst]
// ---------------------------------------------------------------------------
__global__ void mgcn_edge_exp(const int* __restrict__ dst, const float* __restrict__ m,
                              float* __restrict__ ebuf, float* __restrict__ denom, int E) {
    int i = blockIdx.x * blockDim.x + threadIdx.x;
    if (i >= E * 2) return;
    int e = i >> 1, hd = i & 1;
    int d = dst[e];
    float x = __expf(ebuf[i] - m[d * 2 + hd]);
    ebuf[i] = x;
    atomicAdd(&denom[d * 2 + hd], x);
}

// ---------------------------------------------------------------------------
// pass 3: wave per edge; lane handles 4 contiguous columns of the 128-wide
// (H*D) row. Coalesced float4 gather of h[src], 4 f32 atomic adds to agg[dst].
// ---------------------------------------------------------------------------
__global__ void mgcn_edge_agg(const float* __restrict__ ebuf, const float* __restrict__ denom,
                              const int* __restrict__ src, const int* __restrict__ dst,
                              const float* __restrict__ h, float* __restrict__ agg, int E) {
    int gid  = blockIdx.x * blockDim.x + threadIdx.x;
    int edge = gid >> 5;
    if (edge >= E) return;
    int lane = threadIdx.x & 31;
    int s = src[edge], d = dst[edge];
    int c = lane << 2;                    // column 0..124
    int head = c >> 6;                    // 0 or 1
    float a = ebuf[edge * 2 + head] / denom[d * 2 + head];
    const float4 v = *(const float4*)(h + (size_t)s * 128 + c);
    float* out = agg + (size_t)d * 128 + c;
    atomicAdd(out + 0, a * v.x);
    atomicAdd(out + 1, a * v.y);
    atomicAdd(out + 2, a * v.z);
    atomicAdd(out + 3, a * v.w);
}

// ---------------------------------------------------------------------------
// bias add + optional ELU + head mean -> hm[N x 64]
// ---------------------------------------------------------------------------
__global__ void mgcn_post(const float* __restrict__ agg, const float* __restrict__ b,
                          float* __restrict__ hm, int N, int do_elu) {
    int i = blockIdx.x * blockDim.x + threadIdx.x;
    if (i >= N * 64) return;
    int n = i >> 6, d = i & 63;
    float v0 = agg[(size_t)n * 128 + d]      + b[d];
    float v1 = agg[(size_t)n * 128 + 64 + d] + b[64 + d];
    if (do_elu) {
        v0 = (v0 > 0.f) ? v0 : (__expf(v0) - 1.f);
        v1 = (v1 > 0.f) ? v1 : (__expf(v1) - 1.f);
    }
    hm[i] = 0.5f * (v0 + v1);
}

// ---------------------------------------------------------------------------
// z = sigmoid(g + fc1_b + fc2_b + gbias); feats_out = z*hm + (1-z)*feats_in
// ---------------------------------------------------------------------------
__global__ void mgcn_gate(const float* __restrict__ g, const float* __restrict__ fc1_b,
                          const float* __restrict__ fc2_b, const float* __restrict__ gbias,
                          const float* __restrict__ hm, const float* __restrict__ feats_in,
                          float* __restrict__ feats_out, int N) {
    int i = blockIdx.x * blockDim.x + threadIdx.x;
    if (i >= N * 64) return;
    int d = i & 63;
    float zl = g[i] + fc1_b[d] + fc2_b[d] + gbias[d];
    float z  = 1.f / (1.f + __expf(-zl));
    feats_out[i] = z * hm[i] + (1.f - z) * feats_in[i];
}

// ---------------------------------------------------------------------------
extern "C" void kernel_launch(void* const* d_in, const int* in_sizes, int n_in,
                              void* d_out, int out_size, void* d_ws, size_t ws_size,
                              hipStream_t stream) {
    const float* features = (const float*)d_in[0];
    const int*   src      = (const int*)d_in[1];
    const int*   dst      = (const int*)d_in[2];
    const int N = in_sizes[0] / 64;
    const int E = in_sizes[1];

    const float* W[3]  = {(const float*)d_in[3],  (const float*)d_in[7],  (const float*)d_in[11]};
    const float* al[3] = {(const float*)d_in[4],  (const float*)d_in[8],  (const float*)d_in[12]};
    const float* ar[3] = {(const float*)d_in[5],  (const float*)d_in[9],  (const float*)d_in[13]};
    const float* bb[3] = {(const float*)d_in[6],  (const float*)d_in[10], (const float*)d_in[14]};
    const float* fc1_w = (const float*)d_in[15];
    const float* fc1_b = (const float*)d_in[16];
    const float* fc2_w = (const float*)d_in[17];
    const float* fc2_b = (const float*)d_in[18];
    const float* gbias = (const float*)d_in[19];

    float* ws = (float*)d_ws;
    float* h     = ws; ws += (size_t)N * 128;
    float* agg   = ws; ws += (size_t)N * 128;
    float* el    = ws; ws += (size_t)N * 2;
    float* er    = ws; ws += (size_t)N * 2;
    float* mx    = ws; ws += (size_t)N * 2;
    float* denom = ws; ws += (size_t)N * 2;
    float* ebuf  = ws; ws += (size_t)E * 2;
    float* feats = ws; ws += (size_t)N * 64;
    float* hm    = ws; ws += (size_t)N * 64;
    float* gbuf  = ws; ws += (size_t)N * 64;
    // packed f16 weight fragments (all offsets are 32B aligned)
    _Float16* Wp[3];
    Wp[0] = (_Float16*)ws;           ws += 4096;   // 128*64 f16 = 4096 floats
    Wp[1] = (_Float16*)ws;           ws += 4096;
    Wp[2] = (_Float16*)ws;           ws += 4096;
    _Float16* fc1p = (_Float16*)ws;  ws += 2048;   // 64*64 f16
    _Float16* fc2p = (_Float16*)ws;  ws += 2048;

    const int TB = 256;
    const int mtiles = (N + 15) / 16;

    // Pack all weight matrices into WMMA B-fragment f16 layout (once per call)
    for (int L = 0; L < 3; ++L)
        mgcn_pack_b<<<2, TB, 0, stream>>>(W[L], Wp[L], 128);     // 512 threads
    mgcn_pack_b<<<1, TB, 0, stream>>>(fc1_w, fc1p, 64);          // 256 threads
    mgcn_pack_b<<<1, TB, 0, stream>>>(fc2_w, fc2p, 64);

    for (int L = 0; L < 3; ++L) {
        const float* x = (L == 0) ? features : feats;

        // h = x @ W[L]   (N x 64) @ (64 x 128)
        {
            int waves  = mtiles * (128 / 16);
            int blocks = (waves * 32 + TB - 1) / TB;
            mgcn_gemm_wmma<<<blocks, TB, 0, stream>>>(x, Wp[L], h, N, 128, 0);
        }
        mgcn_el_er<<<(N * 2 + TB - 1) / TB, TB, 0, stream>>>(h, al[L], ar[L], el, er, N);
        mgcn_init<<<(N * 128 + TB - 1) / TB, TB, 0, stream>>>(agg, mx, denom, N);
        mgcn_edge_max<<<(E * 2 + TB - 1) / TB, TB, 0, stream>>>(el, er, src, dst, ebuf, mx, E);
        mgcn_edge_exp<<<(E * 2 + TB - 1) / TB, TB, 0, stream>>>(dst, mx, ebuf, denom, E);
        {
            long long thr = (long long)E * 32;
            int blocks = (int)((thr + TB - 1) / TB);
            mgcn_edge_agg<<<blocks, TB, 0, stream>>>(ebuf, denom, src, dst, h, agg, E);
        }
        float* post_out = (L == 0) ? feats : hm;
        mgcn_post<<<(N * 64 + TB - 1) / TB, TB, 0, stream>>>(agg, bb[L], post_out, N, (L < 2) ? 1 : 0);

        if (L > 0) {
            int waves  = mtiles * (64 / 16);
            int blocks = (waves * 32 + TB - 1) / TB;
            mgcn_gemm_wmma<<<blocks, TB, 0, stream>>>(hm,    fc1p, gbuf, N, 64, 0);
            mgcn_gemm_wmma<<<blocks, TB, 0, stream>>>(feats, fc2p, gbuf, N, 64, 1);
            float* fout = (L == 2) ? (float*)d_out : feats;
            mgcn_gate<<<(N * 64 + TB - 1) / TB, TB, 0, stream>>>(
                gbuf, fc1_b, fc2_b, gbias, hm, feats, fout, N);
        }
    }
}
